// T5Attention_89610197663798
// MI455X (gfx1250) — matrix operand order repaired and checked by reading, not compiled, not run on hardware
//
#include <hip/hip_runtime.h>
#include <cstdint>

// ---------------- types ----------------
typedef __attribute__((ext_vector_type(16))) __bf16    v16bf;
typedef __attribute__((ext_vector_type(8)))  float     v8f;
typedef __attribute__((ext_vector_type(4)))  unsigned  v4u;   // POD 16B vector

union Frag {            // one WMMA 16x32 (A) or 32x16 (B) bf16 operand: 8 VGPRs
    v16bf v;
    v4u   q[2];
    unsigned u[8];
};

// global (addrspace 1) access helpers: force global_load/store over flat
typedef const __attribute__((address_space(1))) v4u       GC4;
typedef       __attribute__((address_space(1))) uint16_t  GU16;
typedef       __attribute__((address_space(1))) float     GF32;

__device__ __forceinline__ v4u gload4(const void* p, int idx16B) {
    return ((GC4*)p)[idx16B];
}

__device__ __forceinline__ uint16_t f2bf(float f) {
    unsigned u = __builtin_bit_cast(unsigned, f);
    unsigned r = (u + 0x7FFFu + ((u >> 16) & 1u)) >> 16;   // RNE
    return (uint16_t)r;
}

__device__ __forceinline__ v8f wmma_bf16(const Frag& a, const Frag& b, v8f c) {
    return __builtin_amdgcn_wmma_f32_16x16x32_bf16(
        false, a.v, false, b.v, (short)0, c, false, false);
}

// ---------------- constants ----------------
constexpr int BATCH = 4, SEQ = 1024, DMODEL = 1024, HEADS = 16, DKV = 64;
constexpr int MROWS = BATCH * SEQ;          // 4096
constexpr int NBUCKETS = 32;
constexpr int BIAS_W = 2 * SEQ - 1;         // 2047 deltas

// workspace offsets (bytes)
constexpr size_t OFF_XBF  = 0;                          // 4096x1024 bf16
constexpr size_t OFF_WQ   = OFF_XBF  + (size_t)MROWS * DMODEL * 2;
constexpr size_t OFF_WK   = OFF_WQ   + (size_t)DMODEL * DMODEL * 2;
constexpr size_t OFF_WV   = OFF_WK   + (size_t)DMODEL * DMODEL * 2;
constexpr size_t OFF_WO   = OFF_WV   + (size_t)DMODEL * DMODEL * 2;
constexpr size_t OFF_Q    = OFF_WO   + (size_t)DMODEL * DMODEL * 2;
constexpr size_t OFF_K    = OFF_Q    + (size_t)MROWS * DMODEL * 2;
constexpr size_t OFF_VT   = OFF_K    + (size_t)MROWS * DMODEL * 2;
constexpr size_t OFF_CTX  = OFF_VT   + (size_t)MROWS * DMODEL * 2;
constexpr size_t OFF_BIAS = OFF_CTX  + (size_t)MROWS * DMODEL * 2;
constexpr size_t WS_NEED  = OFF_BIAS + (size_t)HEADS * BIAS_W * 4;

// ---------------- prep kernels ----------------
__global__ void t5_conv_bf16(const float* __restrict__ x,
                             uint16_t* __restrict__ y, int n) {
    int i = blockIdx.x * blockDim.x + threadIdx.x;
    if (i < n) ((GU16*)y)[i] = f2bf(x[i]);
}

// Pack fp32 W[K=1024][N=1024] row-major into WMMA B-fragment layout (bf16).
// Fragment (kb, nb): 32x16 tile; lane l<16 holds col nb*16+l, k=kb*32+0..15;
// lane l>=16 holds col nb*16+l-16, k=kb*32+16..31; pairs packed low/high.
__global__ void t5_pack_w(const float* __restrict__ W, unsigned* __restrict__ out) {
    int tid  = blockIdx.x * blockDim.x + threadIdx.x;   // 65536 threads
    int lane = tid & 31;
    int nb   = (tid >> 5) & 63;
    int kb   = tid >> 11;                               // 0..31
    int col  = nb * 16 + (lane & 15);
    int k0   = kb * 32 + ((lane & 16) ? 16 : 0);
    unsigned* dst = out + ((size_t)(kb * 64 + nb) * 32 + lane) * 8;
#pragma unroll
    for (int j = 0; j < 8; ++j) {
        float lo = W[(k0 + 2 * j) * DMODEL + col];
        float hi = W[(k0 + 2 * j + 1) * DMODEL + col];
        dst[j] = (unsigned)f2bf(lo) | ((unsigned)f2bf(hi) << 16);
    }
}

// dbias[h][delta+1023] = rel_bias_table[bucket(delta)][h]
__global__ void t5_bias(const float* __restrict__ table, float* __restrict__ dbias) {
    int idx = blockIdx.x * blockDim.x + threadIdx.x;
    if (idx >= BIAS_W) return;
    int rel = idx - (SEQ - 1);
    int nb = NBUCKETS / 2;                 // 16
    int bucket = (rel > 0) ? nb : 0;
    int ap = rel < 0 ? -rel : rel;
    int max_exact = nb / 2;                // 8
    if (ap < max_exact) {
        bucket += ap;
    } else {
        float rp = (float)(ap < 1 ? 1 : ap);
        int large = max_exact +
            (int)(__logf(rp / (float)max_exact) / __logf(128.0f / (float)max_exact) *
                  (float)(nb - max_exact));
        if (large > nb - 1) large = nb - 1;
        bucket += large;
    }
#pragma unroll
    for (int h = 0; h < HEADS; ++h)
        dbias[h * BIAS_W + idx] = table[bucket * HEADS + h];
}

// ---------------- bf16 WMMA GEMM: C[4096x1024] = A[4096x1024] * B ----------------
// block = 128 threads (4 waves); wave tile 64x64; block tile 64M x 256N.
// mode 0: bf16 row-major out; 1: bf16 V-transposed ([b][h][dv][s]); 2: fp32 row-major.
__global__ __launch_bounds__(128, 1)
void t5_gemm(const uint16_t* __restrict__ A,
             const unsigned* __restrict__ Bp,
             void* __restrict__ out, int mode) {
    const int lane = threadIdx.x & 31;
    const int w    = threadIdx.x >> 5;
    const int ln15 = lane & 15;
    const int hi   = (lane & 16) ? 1 : 0;
    const int m0 = blockIdx.y * 64;
    const int n0 = blockIdx.x * 256 + w * 64;

    v8f acc[4][4];
#pragma unroll
    for (int i = 0; i < 4; ++i)
#pragma unroll
        for (int j = 0; j < 4; ++j) acc[i][j] = {};

    // per-lane streaming pointers (incremented each K-step; no re-derivation)
    const uint16_t* ap[4];
#pragma unroll
    for (int mt = 0; mt < 4; ++mt)
        ap[mt] = A + (size_t)(m0 + mt * 16 + ln15) * DMODEL + (hi ? 16 : 0);

    const unsigned* bptr[4];
#pragma unroll
    for (int nt = 0; nt < 4; ++nt) {
        int nbg = blockIdx.x * 16 + w * 4 + nt;
        bptr[nt] = Bp + ((size_t)nbg * 32 + lane) * 8;
    }
    constexpr size_t BSTRIDE = (size_t)64 * 32 * 8;     // unsigned per K-step

    for (int kb = 0; kb < DMODEL / 32; ++kb) {
        Frag af[4];
#pragma unroll
        for (int mt = 0; mt < 4; ++mt) {
            af[mt].q[0] = gload4(ap[mt], 0);     // k .. k+7
            af[mt].q[1] = gload4(ap[mt], 1);     // k+8 .. k+15
            __builtin_prefetch(ap[mt] + 32, 0, 3);   // next K-step A line
            ap[mt] += 32;
        }
#pragma unroll
        for (int nt = 0; nt < 4; ++nt) {
            Frag bf_;
            bf_.q[0] = gload4(bptr[nt], 0);
            bf_.q[1] = gload4(bptr[nt], 1);
            __builtin_prefetch(bptr[nt] + BSTRIDE, 0, 3);  // next K-step B frag
            bptr[nt] += BSTRIDE;
#pragma unroll
            for (int mt = 0; mt < 4; ++mt)
                acc[mt][nt] = wmma_bf16(af[mt], bf_, acc[mt][nt]);
        }
    }

#pragma unroll
    for (int mt = 0; mt < 4; ++mt)
#pragma unroll
        for (int nt = 0; nt < 4; ++nt) {
            int n = n0 + nt * 16 + ln15;
#pragma unroll
            for (int r = 0; r < 8; ++r) {
                int m = m0 + mt * 16 + r + hi * 8;
                float vv = acc[mt][nt][r];
                if (mode == 0) {
                    ((GU16*)out)[(size_t)m * DMODEL + n] = f2bf(vv);
                } else if (mode == 1) {
                    int b = m >> 10, s = m & 1023, h = n >> 6, dv = n & 63;
                    ((GU16*)out)[(((size_t)(b * HEADS + h) * DKV) + dv) * SEQ + s] = f2bf(vv);
                } else {
                    ((GF32*)out)[(size_t)m * DMODEL + n] = vv;
                }
            }
        }
}

// ---------------- flash attention: per (b,h,64-row q-block) ----------------
// 4 waves per block; wave owns 16 q rows; streams 64-key blocks with online softmax.
__global__ __launch_bounds__(128, 1)
void t5_attn(const uint16_t* __restrict__ Q,
             const uint16_t* __restrict__ K,
             const uint16_t* __restrict__ VT,
             const float* __restrict__ dbias,
             uint16_t* __restrict__ CTX) {
    __shared__ float    bsh[BIAS_W];
    __shared__ uint16_t pst[4 * 16 * 72];   // per-wave 16x64 P tile, row stride 72 (pad)

    const int tid = threadIdx.x, lane = tid & 31, w = tid >> 5;
    const int ln15 = lane & 15;
    const int hi   = (lane & 16) ? 1 : 0;
    const int qb = blockIdx.x, h = blockIdx.y, b = blockIdx.z;

    for (int i = tid; i < BIAS_W; i += 128) bsh[i] = dbias[h * BIAS_W + i];
    __syncthreads();

    const int q0 = qb * 64 + w * 16;

    Frag qf[2];
#pragma unroll
    for (int dc = 0; dc < 2; ++dc) {
        int row = b * SEQ + q0 + ln15;
        int c0  = h * DKV + dc * 32 + (hi ? 16 : 0);
        const uint16_t* p = Q + (size_t)row * DMODEL + c0;
        qf[dc].q[0] = gload4(p, 0);
        qf[dc].q[1] = gload4(p, 1);
    }

    // streaming pointers: K rows for this lane (advance 64 rows per key-block),
    // VT rows for this lane's dv column (advance 64 keys per key-block)
    const uint16_t* kp = K + (size_t)(b * SEQ + ln15) * DMODEL + h * DKV + (hi ? 16 : 0);
    const uint16_t* vp = VT + ((size_t)(b * HEADS + h) * DKV + ln15) * SEQ + (hi ? 16 : 0);

    float mrow[8], lrow[8];
    v8f oacc[4];
#pragma unroll
    for (int r = 0; r < 8; ++r) { mrow[r] = -1e30f; lrow[r] = 0.f; }
#pragma unroll
    for (int nt = 0; nt < 4; ++nt) oacc[nt] = {};

    const int biasBase0 = (SEQ - 1) + ln15 - (q0 + hi * 8);  // + kbb + t*16 - r

    for (int kblk = 0; kblk < SEQ / 64; ++kblk) {
        const int kbb = kblk * 64;

        // ---- scores S = (Q K^T) * 0.125 + bias  (16 x 64) ----
        v8f sacc[4];
#pragma unroll
        for (int t = 0; t < 4; ++t) sacc[t] = {};
#pragma unroll
        for (int t = 0; t < 4; ++t) {
            const uint16_t* kr = kp + (size_t)(kbb + t * 16) * DMODEL;
            Frag kf0, kf1;
            kf0.q[0] = gload4(kr, 0);  kf0.q[1] = gload4(kr, 1);
            kf1.q[0] = gload4(kr, 2);  kf1.q[1] = gload4(kr, 3);
            sacc[t] = wmma_bf16(qf[0], kf0, sacc[t]);
            sacc[t] = wmma_bf16(qf[1], kf1, sacc[t]);
        }

        // fold scale + bias in place
#pragma unroll
        for (int t = 0; t < 4; ++t) {
            int bb = biasBase0 + kbb + t * 16;
#pragma unroll
            for (int r = 0; r < 8; ++r)
                sacc[t][r] = sacc[t][r] * 0.125f + bsh[bb - r];
        }

        // ---- online softmax ----
        float mx[8];
#pragma unroll
        for (int r = 0; r < 8; ++r) {
            float m = fmaxf(fmaxf(sacc[0][r], sacc[1][r]),
                            fmaxf(sacc[2][r], sacc[3][r]));
            mx[r] = m;
        }
#pragma unroll
        for (int off = 1; off <= 8; off <<= 1)
#pragma unroll
            for (int r = 0; r < 8; ++r)
                mx[r] = fmaxf(mx[r], __shfl_xor(mx[r], off, 32));

        float corr[8];
#pragma unroll
        for (int r = 0; r < 8; ++r) {
            float mn = fmaxf(mrow[r], mx[r]);
            corr[r] = __expf(mrow[r] - mn);
            mrow[r] = mn;
        }

        float sum[8];
#pragma unroll
        for (int r = 0; r < 8; ++r) sum[r] = 0.f;
#pragma unroll
        for (int t = 0; t < 4; ++t)
#pragma unroll
            for (int r = 0; r < 8; ++r) {
                float pv = __expf(sacc[t][r] - mrow[r]);
                sum[r] += pv;
                pst[(size_t)(w * 16 + r + hi * 8) * 72 + t * 16 + ln15] = f2bf(pv);
            }
#pragma unroll
        for (int off = 1; off <= 8; off <<= 1)
#pragma unroll
            for (int r = 0; r < 8; ++r)
                sum[r] += __shfl_xor(sum[r], off, 32);
#pragma unroll
        for (int r = 0; r < 8; ++r) lrow[r] = lrow[r] * corr[r] + sum[r];
#pragma unroll
        for (int nt = 0; nt < 4; ++nt)
#pragma unroll
            for (int r = 0; r < 8; ++r) oacc[nt][r] *= corr[r];

        __syncthreads();   // P staged to LDS (C-layout) -> reload in A-layout
        Frag pf[2];
#pragma unroll
        for (int dc = 0; dc < 2; ++dc) {
            const v4u* lp = reinterpret_cast<const v4u*>(
                pst + (size_t)(w * 16 + ln15) * 72 + dc * 32 + (hi ? 16 : 0));
            pf[dc].q[0] = lp[0];
            pf[dc].q[1] = lp[1];
        }
        __syncthreads();

        // ---- O += P * V ----
#pragma unroll
        for (int nt = 0; nt < 4; ++nt) {
            const uint16_t* vr = vp + (size_t)(nt * 16) * SEQ + kbb;
#pragma unroll
            for (int kcb = 0; kcb < 2; ++kcb) {
                Frag vf;
                vf.q[0] = gload4(vr, kcb * 2);
                vf.q[1] = gload4(vr, kcb * 2 + 1);
                oacc[nt] = wmma_bf16(pf[kcb], vf, oacc[nt]);
            }
        }
    }

    // ---- normalize + store ctx (bf16 row-major [4096 x 1024]) ----
    float rinv[8];
#pragma unroll
    for (int r = 0; r < 8; ++r) rinv[r] = __frcp_rn(lrow[r]);
#pragma unroll
    for (int nt = 0; nt < 4; ++nt)
#pragma unroll
        for (int r = 0; r < 8; ++r) {
            float vv = oacc[nt][r] * rinv[r];
            int row = b * SEQ + q0 + r + hi * 8;
            int col = h * DKV + nt * 16 + ln15;
            ((GU16*)CTX)[(size_t)row * DMODEL + col] = f2bf(vv);
        }
}

// ---------------- host launcher ----------------
extern "C" void kernel_launch(void* const* d_in, const int* in_sizes, int n_in,
                              void* d_out, int out_size, void* d_ws, size_t ws_size,
                              hipStream_t stream) {
    if (ws_size < WS_NEED) return;
    const float* X     = (const float*)d_in[0];
    const float* Wq    = (const float*)d_in[1];
    const float* Wk    = (const float*)d_in[2];
    const float* Wv    = (const float*)d_in[3];
    const float* Wo    = (const float*)d_in[4];
    const float* table = (const float*)d_in[5];

    char* ws = (char*)d_ws;
    uint16_t* Xbf  = (uint16_t*)(ws + OFF_XBF);
    unsigned* WqP  = (unsigned*)(ws + OFF_WQ);
    unsigned* WkP  = (unsigned*)(ws + OFF_WK);
    unsigned* WvP  = (unsigned*)(ws + OFF_WV);
    unsigned* WoP  = (unsigned*)(ws + OFF_WO);
    uint16_t* Qbf  = (uint16_t*)(ws + OFF_Q);
    uint16_t* Kbf  = (uint16_t*)(ws + OFF_K);
    uint16_t* VTbf = (uint16_t*)(ws + OFF_VT);
    uint16_t* Ctx  = (uint16_t*)(ws + OFF_CTX);
    float*    Bias = (float*)(ws + OFF_BIAS);

    // prep
    {
        int n = MROWS * DMODEL;
        t5_conv_bf16<<<(n + 255) / 256, 256, 0, stream>>>(X, Xbf, n);
    }
    t5_pack_w<<<256, 256, 0, stream>>>(Wq, WqP);
    t5_pack_w<<<256, 256, 0, stream>>>(Wk, WkP);
    t5_pack_w<<<256, 256, 0, stream>>>(Wv, WvP);
    t5_bias<<<(BIAS_W + 255) / 256, 256, 0, stream>>>(table, Bias);
    t5_pack_w<<<256, 256, 0, stream>>>(Wo, WoP);

    // projections
    dim3 ggrid(DMODEL / 256, MROWS / 64);
    t5_gemm<<<ggrid, 128, 0, stream>>>(Xbf, WqP, (void*)Qbf, 0);
    t5_gemm<<<ggrid, 128, 0, stream>>>(Xbf, WkP, (void*)Kbf, 0);
    t5_gemm<<<ggrid, 128, 0, stream>>>(Xbf, WvP, (void*)VTbf, 1);

    // attention
    dim3 agrid(SEQ / 64, HEADS, BATCH);
    t5_attn<<<agrid, 128, 0, stream>>>(Qbf, Kbf, VTbf, Bias, Ctx);

    // output projection (fp32 out)
    t5_gemm<<<ggrid, 128, 0, stream>>>(Ctx, WoP, d_out, 2);
}
